// NodeRegressor_59708635349189
// MI455X (gfx1250) — matrix operand order repaired and checked by reading, compiled, not run on hardware
//
#include <hip/hip_runtime.h>
#include <hip/hip_bf16.h>

typedef __attribute__((ext_vector_type(16))) _Float16 v16h;
typedef __attribute__((ext_vector_type(8)))  float    v8f;

#define NNODES 100000
#define NEDGES 1200000
#define FCONT  16
#define EMBD   8
#define HID    64
#define NLAY   3
#define TPB    256

// ---------------------------------------------------------------- utilities
__global__ void zero_f32_kernel(float* __restrict__ p, int n) {
  int i = blockIdx.x * blockDim.x + threadIdx.x;
  if (i < n) p[i] = 0.0f;
}

__global__ void count_deg_kernel(const int* __restrict__ dst, float* __restrict__ deg, int E) {
  int e = blockIdx.x * blockDim.x + threadIdx.x;
  if (e < E) atomicAdd(&deg[dst[e]], 1.0f);
}

__global__ void finalize_deg_kernel(float* __restrict__ deg, int n) {
  int i = blockIdx.x * blockDim.x + threadIdx.x;
  if (i < n) deg[i] = rsqrtf(1.0f + deg[i]);   // deg_isq
}

// pack concat(x_cont[16], emb[func_type][8]) into K=32 padded rows (cols 24..31 = 0)
__global__ void prep_input_kernel(const float* __restrict__ xc, const int* __restrict__ ft,
                                  const float* __restrict__ emb, float* __restrict__ xin, int n) {
  int t = blockIdx.x * blockDim.x + threadIdx.x;
  if (t >= n * 32) return;
  int node = t >> 5, f = t & 31;
  float v = 0.0f;
  if (f < FCONT)             v = xc[node * FCONT + f];
  else if (f < FCONT + EMBD) v = emb[ft[node] * EMBD + (f - FCONT)];
  xin[t] = v;
}

// ---------------------------------------------------------------- WMMA GEMM
// Y[nrows,64] = act(X[nrows,K] @ W[Kw,64] (+bias)).  K in {32,64} (compile-time),
// W zero-padded to K rows.  Weights are staged in LDS pre-converted to f16 and
// pre-swizzled into the per-lane B-fragment layout, so each wave builds all
// K/32 x 4 B fragments with contiguous ds_load_b128s exactly once, then sweeps
// TILES_PER_WAVE 16-row tiles reusing them.  f16 multiply, f32 accumulate.
template <int K>
__global__ void __launch_bounds__(256)
gemm_wmma_kernel(const float* __restrict__ X, const float* __restrict__ W,
                 const float* __restrict__ bias, float* __restrict__ Y,
                 int ntiles, int Kw, int do_relu) {
  constexpr int KSTEPS = K / 32;
  constexpr int TILES_PER_WAVE = 2;

  // Wh[ks][nb][lane][j]: f16 B-fragment element j of lane `lane` for column
  // block nb, K-step ks.  n = nb*16 + (lane&15); k = ks*32 + (lane>>4)*16 + j.
  __shared__ alignas(32) _Float16 Wh[KSTEPS * 4 * 32 * 16];
  {
    int tid = threadIdx.x;
    if (tid < KSTEPS * 4 * 32) {
      int lane_s = tid & 31;
      int nb_s   = (tid >> 5) & 3;
      int ks_s   = tid >> 7;
      int n      = nb_s * 16 + (lane_s & 15);
      int kbase  = ks_s * 32 + (lane_s >> 4) * 16;
      #pragma unroll
      for (int j = 0; j < 16; ++j) {
        int k = kbase + j;
        Wh[tid * 16 + j] = (k < Kw) ? (_Float16)W[k * HID + n] : (_Float16)0.0f;
      }
    }
  }
  __syncthreads();

  int lane = threadIdx.x & 31;
  int wave = threadIdx.x >> 5;
  int m    = lane & 15;   // A row / B,C column within the 16-block
  int hi   = lane >> 4;   // K-half selector (contiguous k = 16*hi + j, same for A and B)

  // hoist all B fragments into registers (two ds_load_b128 each, no cvt)
  v16h bfrag[KSTEPS][4];
  #pragma unroll
  for (int ks = 0; ks < KSTEPS; ++ks)
    #pragma unroll
    for (int nb = 0; nb < 4; ++nb)
      bfrag[ks][nb] = *(const v16h*)&Wh[((ks * 4 + nb) * 32 + lane) * 16];

  int tile0 = (blockIdx.x * 8 + wave) * TILES_PER_WAVE;

  for (int t = 0; t < TILES_PER_WAVE; ++t) {
    int tile = tile0 + t;
    if (tile >= ntiles) return;           // wave-uniform; EXEC stays all-ones
    int row0 = tile * 16;

    v8f acc[4];
    #pragma unroll
    for (int nb = 0; nb < 4; ++nb)
      #pragma unroll
      for (int r = 0; r < 8; ++r) acc[nb][r] = 0.0f;

    #pragma unroll
    for (int ks = 0; ks < KSTEPS; ++ks) {
      // A fragment 16x32 f16: lanes 0-15 K={0..15}; lanes 16-31 K={16..31}
      v16h a;
      const float* xrow = X + (size_t)(row0 + m) * K + ks * 32 + hi * 16;
      #pragma unroll
      for (int j = 0; j < 8; ++j) a[j]     = (_Float16)xrow[j];
      #pragma unroll
      for (int j = 0; j < 8; ++j) a[8 + j] = (_Float16)xrow[8 + j];

      #pragma unroll
      for (int nb = 0; nb < 4; ++nb)
        acc[nb] = __builtin_amdgcn_wmma_f32_16x16x32_f16(
            false, a, false, bfrag[ks][nb], (short)0, acc[nb], false, false);
    }

    // C/D layout: VGPR r -> row = r + hi*8, col = lane&15 within the 16-block
    #pragma unroll
    for (int nb = 0; nb < 4; ++nb) {
      int col = nb * 16 + m;
      float bv = bias ? bias[col] : 0.0f;
      #pragma unroll
      for (int r = 0; r < 8; ++r) {
        int row = row0 + r + hi * 8;
        float v = acc[nb][r] + bv;
        if (do_relu) v = fmaxf(v, 0.0f);
        Y[(size_t)row * HID + col] = v;
      }
    }
  }
}

// ---------------------------------------------------------------- GCN aggregation
// thread t = (edge e, 4-feature group): float4 gather of h[src], 4 f32 atomics
__global__ void edge_scatter_kernel(const int* __restrict__ src, const int* __restrict__ dst,
                                    const float* __restrict__ deg_isq,
                                    const float* __restrict__ h, float* __restrict__ agg, int E) {
  int t = blockIdx.x * blockDim.x + threadIdx.x;
  if (t >= E * (HID / 4)) return;
  int e = t >> 4;
  int f = (t & 15) * 4;
  int s = src[e], d = dst[e];
  float nrm = deg_isq[s] * deg_isq[d];
  float4 hv = *(const float4*)&h[(size_t)s * HID + f];
  float* ap = &agg[(size_t)d * HID + f];
  atomicAdd(ap + 0, nrm * hv.x);
  atomicAdd(ap + 1, nrm * hv.y);
  atomicAdd(ap + 2, nrm * hv.z);
  atomicAdd(ap + 3, nrm * hv.w);
}

// x = relu(agg + d^-1 * h + b)   (self-loop contribution + bias + activation)
__global__ void combine_relu_kernel(const float* __restrict__ agg, const float* __restrict__ h,
                                    const float* __restrict__ deg_isq,
                                    const float* __restrict__ b, float* __restrict__ x, int n) {
  int t = blockIdx.x * blockDim.x + threadIdx.x;
  if (t >= n * (HID / 4)) return;
  int node = t >> 4;
  int f    = (t & 15) * 4;
  float di = deg_isq[node];
  float d2 = di * di;
  float4 av = *(const float4*)&agg[(size_t)node * HID + f];
  float4 hv = *(const float4*)&h[(size_t)node * HID + f];
  float4 bv = *(const float4*)&b[f];
  float4 r;
  r.x = fmaxf(av.x + d2 * hv.x + bv.x, 0.0f);
  r.y = fmaxf(av.y + d2 * hv.y + bv.y, 0.0f);
  r.z = fmaxf(av.z + d2 * hv.z + bv.z, 0.0f);
  r.w = fmaxf(av.w + d2 * hv.w + bv.w, 0.0f);
  *(float4*)&x[(size_t)node * HID + f] = r;
}

// out[node] = h[node,:] . W_o2 + b_o2  (wave32 per node, butterfly reduce)
__global__ void final_dot_kernel(const float* __restrict__ h, const float* __restrict__ w2,
                                 const float* __restrict__ b2, float* __restrict__ out, int n) {
  int lane = threadIdx.x & 31;
  int node = (blockIdx.x * blockDim.x + threadIdx.x) >> 5;
  if (node >= n) return;
  const float* row = h + (size_t)node * HID;
  float v = row[lane] * w2[lane] + row[lane + 32] * w2[lane + 32];
  #pragma unroll
  for (int off = 16; off > 0; off >>= 1) v += __shfl_xor(v, off, 32);
  if (lane == 0) out[node] = v + b2[0];
}

// ---------------------------------------------------------------- launch
extern "C" void kernel_launch(void* const* d_in, const int* in_sizes, int n_in,
                              void* d_out, int out_size, void* d_ws, size_t ws_size,
                              hipStream_t stream) {
  const float* x_cont    = (const float*)d_in[0];
  const int*   func_type = (const int*)  d_in[1];
  const int*   edge_idx  = (const int*)  d_in[2];
  const float* emb       = (const float*)d_in[3];
  const float* W_in      = (const float*)d_in[4];
  const float* b_in      = (const float*)d_in[5];
  const float* W_g       = (const float*)d_in[6];
  const float* b_g       = (const float*)d_in[7];
  const float* W_o1      = (const float*)d_in[8];
  const float* b_o1      = (const float*)d_in[9];
  const float* W_o2      = (const float*)d_in[10];
  const float* b_o2      = (const float*)d_in[11];
  float* out = (float*)d_out;

  const int* src = edge_idx;            // edge_index[0]
  const int* dst = edge_idx + NEDGES;   // edge_index[1]

  // workspace (floats): deg_isq[N] | xin[N*32] | x[N*64] | h[N*64] | agg[N*64]
  float* deg  = (float*)d_ws;
  float* xin  = deg  + NNODES;
  float* xbuf = xin  + (size_t)NNODES * 32;
  float* hbuf = xbuf + (size_t)NNODES * HID;
  float* agg  = hbuf + (size_t)NNODES * HID;

  auto cdiv = [](int a, int b) { return (a + b - 1) / b; };
  const int ntiles  = cdiv(NNODES, 16);            // 6250 (exact)
  const int gblocks = cdiv(ntiles, 8 * 2);         // 8 waves x 2 tiles per block

  // degrees -> deg_isq
  zero_f32_kernel<<<cdiv(NNODES, TPB), TPB, 0, stream>>>(deg, NNODES);
  count_deg_kernel<<<cdiv(NEDGES, TPB), TPB, 0, stream>>>(dst, deg, NEDGES);
  finalize_deg_kernel<<<cdiv(NNODES, TPB), TPB, 0, stream>>>(deg, NNODES);

  // input embedding concat + input layer (K padded 24 -> 32)
  prep_input_kernel<<<cdiv(NNODES * 32, TPB), TPB, 0, stream>>>(x_cont, func_type, emb, xin, NNODES);
  gemm_wmma_kernel<32><<<gblocks, 256, 0, stream>>>(xin, W_in, b_in, xbuf, ntiles, FCONT + EMBD, 1);

  // 3 GCN layers
  for (int l = 0; l < NLAY; ++l) {
    gemm_wmma_kernel<64><<<gblocks, 256, 0, stream>>>(xbuf, W_g + (size_t)l * HID * HID, nullptr,
                                                      hbuf, ntiles, HID, 0);
    zero_f32_kernel<<<cdiv(NNODES * HID, TPB), TPB, 0, stream>>>(agg, NNODES * HID);
    edge_scatter_kernel<<<cdiv(NEDGES * (HID / 4), TPB), TPB, 0, stream>>>(src, dst, deg, hbuf, agg, NEDGES);
    combine_relu_kernel<<<cdiv(NNODES * (HID / 4), TPB), TPB, 0, stream>>>(agg, hbuf, deg,
                                                                           b_g + (size_t)l * HID, xbuf, NNODES);
  }

  // output head
  gemm_wmma_kernel<64><<<gblocks, 256, 0, stream>>>(xbuf, W_o1, b_o1, hbuf, NNODES / 16, HID, 1);
  final_dot_kernel<<<cdiv(NNODES * 32, TPB), TPB, 0, stream>>>(hbuf, W_o2, b_o2, out, NNODES);
}
